// sMoCo_1821066133823
// MI455X (gfx1250) — matrix operand order repaired and checked by reading, not compiled, not run on hardware
//
#include <hip/hip_runtime.h>
#include <hip/hip_bf16.h>

// ---------------- problem constants ----------------
#define B_    512
#define D_    256
#define KQ    65536
#define NPOS_ 3
#define LROW  (NPOS_ + KQ)            // 65539
static constexpr float T_     = 0.07f;
static constexpr float INV_T  = 1.0f / 0.07f;

// ---------------- vector types ----------------
typedef __attribute__((ext_vector_type(16))) _Float16 v16h;
typedef __attribute__((ext_vector_type(8)))  _Float16 v8h;
typedef __attribute__((ext_vector_type(4)))  _Float16 v4h;
typedef __attribute__((ext_vector_type(8)))  float    v8f;
typedef __attribute__((ext_vector_type(4)))  float    v4f;

// ---------------- workspace layout (bytes) ----------------
static constexpr size_t WS_QNORM = 0;                                   // f32 [512][256]
static constexpr size_t WS_QH    = WS_QNORM + (size_t)B_ * D_ * 4;      // f16 [512][256]
static constexpr size_t WS_QL    = WS_QH    + (size_t)B_ * D_ * 2;      // f16 [512][256]
static constexpr size_t WS_SPOS  = WS_QL    + (size_t)B_ * D_ * 2;      // f32 [512][3]
static constexpr size_t WS_PMIN  = WS_SPOS  + (size_t)B_ * 3  * 4;      // f32 [512]
static constexpr size_t WS_BH    = WS_PMIN  + (size_t)B_ * 4 + 256;     // f16 [65536][256] (16B aligned)
static constexpr size_t WS_BL    = WS_BH    + (size_t)KQ * D_ * 2;      // f16 [65536][256]

// ---------------- output layout (f32 elements) ----------------
static constexpr size_t OUT_LOGITS = 0;
static constexpr size_t OUT_LABELS = (size_t)B_ * LROW;
static constexpr size_t OUT_SCORE  = 2 * (size_t)B_ * LROW;
static constexpr size_t OUT_NEWQ   = OUT_SCORE + (size_t)B_ * 7;

// =====================================================================
// K1: L2-normalize query rows and key rows.
// =====================================================================
__global__ void k_normalize(const float* __restrict__ query,
                            const float* __restrict__ key_emb,
                            float* __restrict__ q_norm,
                            _Float16* __restrict__ qh,
                            _Float16* __restrict__ ql,
                            float* __restrict__ newq)
{
    __shared__ float red[9];
    const int row = blockIdx.x;          // 0..2047
    const int t   = threadIdx.x;         // 0..255 (== D_)
    const bool isq = row < B_;
    const float* src = isq ? (query + (size_t)row * D_)
                           : (key_emb + (size_t)(row - B_) * D_);
    const float v = src[t];

    float s = v * v;
    for (int o = 16; o > 0; o >>= 1) s += __shfl_down(s, o, 32);
    const int lane = t & 31, w = t >> 5;
    if (lane == 0) red[w] = s;
    __syncthreads();
    if (t == 0) {
        float tot = 0.f;
        for (int i = 0; i < 8; ++i) tot += red[i];
        red[8] = 1.0f / fmaxf(sqrtf(tot), 1e-12f);
    }
    __syncthreads();
    const float y = v * red[8];

    if (isq) {
        q_norm[(size_t)row * D_ + t] = y;
        const _Float16 h = (_Float16)y;
        qh[(size_t)row * D_ + t] = h;
        ql[(size_t)row * D_ + t] = (_Float16)(y - (float)h);
    } else {
        newq[(size_t)(KQ - NPOS_ * B_ + (row - B_)) * D_ + t] = y;
    }
}

// =====================================================================
// K2: split queue f32 -> (hi, lo) f16 pair  (error-compensated operand)
// =====================================================================
__global__ void k_split_queue(const float* __restrict__ queue,
                              _Float16* __restrict__ bhv,
                              _Float16* __restrict__ blv)
{
    const size_t n = (size_t)KQ * D_;
    const size_t stride = (size_t)gridDim.x * blockDim.x * 4;
    for (size_t i = ((size_t)blockIdx.x * blockDim.x + threadIdx.x) * 4;
         i < n; i += stride) {
        const float4 v = *(const float4*)(queue + i);
        v4h h, l;
        h[0] = (_Float16)v.x; l[0] = (_Float16)(v.x - (float)h[0]);
        h[1] = (_Float16)v.y; l[1] = (_Float16)(v.y - (float)h[1]);
        h[2] = (_Float16)v.z; l[2] = (_Float16)(v.z - (float)h[2]);
        h[3] = (_Float16)v.w; l[3] = (_Float16)(v.w - (float)h[3]);
        *(v4h*)(bhv + i) = h;
        *(v4h*)(blv + i) = l;
    }
}

// =====================================================================
// K3: positive scores (3 dots per row), logits[:, :3], pos_min
// =====================================================================
__global__ void k_pos(const float* __restrict__ q_norm,
                      const float* __restrict__ newq,
                      float* __restrict__ logits,
                      float* __restrict__ score_pos,
                      float* __restrict__ pos_min)
{
    __shared__ float red[24];
    const int b = blockIdx.x, t = threadIdx.x;
    const float qv = q_norm[(size_t)b * D_ + t];
    const size_t kbase = (size_t)(KQ - NPOS_ * B_ + b * 3) * D_;
    float d0 = qv * newq[kbase + 0 * D_ + t];
    float d1 = qv * newq[kbase + 1 * D_ + t];
    float d2 = qv * newq[kbase + 2 * D_ + t];
    for (int o = 16; o > 0; o >>= 1) {
        d0 += __shfl_down(d0, o, 32);
        d1 += __shfl_down(d1, o, 32);
        d2 += __shfl_down(d2, o, 32);
    }
    const int lane = t & 31, w = t >> 5;
    if (lane == 0) { red[w] = d0; red[8 + w] = d1; red[16 + w] = d2; }
    __syncthreads();
    if (t == 0) {
        float s0 = 0.f, s1 = 0.f, s2 = 0.f;
        for (int i = 0; i < 8; ++i) { s0 += red[i]; s1 += red[8 + i]; s2 += red[16 + i]; }
        float* lrow = logits + (size_t)b * LROW;
        lrow[0] = s0 * INV_T; lrow[1] = s1 * INV_T; lrow[2] = s2 * INV_T;
        score_pos[b * 3 + 0] = s0;
        score_pos[b * 3 + 1] = s1;
        score_pos[b * 3 + 2] = s2;
        pos_min[b] = fminf(s0, fminf(s1, s2));
    }
}

// =====================================================================
// K4: WMMA GEMM  score_neg = q @ queue^T  (split-f16, f32 accumulate)
//   Staging now uses CDNA5 async global->LDS DMA (ASYNCcnt), bypassing
//   VGPRs: global_load_async_to_lds_b128 + s_wait_asynccnt.
// =====================================================================
#define MT   64
#define NT   64
#define STRH 264   // padded LDS row stride in halves (528 B = 132 dwords)

static __device__ __forceinline__ v16h cat16(v8h a, v8h b) {
    return __builtin_shufflevector(a, b, 0, 1, 2, 3, 4, 5, 6, 7,
                                         8, 9, 10, 11, 12, 13, 14, 15);
}
static __device__ __forceinline__ v8f wmma_f16(v16h a, v16h b, v8f c) {
    return __builtin_amdgcn_wmma_f32_16x16x32_f16(false, a, false, b,
                                                  (short)0, c, false, false);
}
// CDNA5 async DMA: 16B global -> LDS, no VGPR round-trip (ASYNCcnt).
static __device__ __forceinline__ void async_copy16(const _Float16* gbase,
                                                    int byteOff,
                                                    unsigned ldsOff) {
    asm volatile("global_load_async_to_lds_b128 %0, %1, %2"
                 :
                 : "v"(ldsOff), "v"(byteOff), "s"(gbase)
                 : "memory");
}

__global__ void __launch_bounds__(256)
k_gemm(const _Float16* __restrict__ qh, const _Float16* __restrict__ ql,
       const _Float16* __restrict__ bhv, const _Float16* __restrict__ blv,
       float* __restrict__ logits)
{
    extern __shared__ _Float16 sm[];
    _Float16* sAh = sm;
    _Float16* sAl = sm + 1 * MT * STRH;
    _Float16* sBh = sm + 2 * MT * STRH;
    _Float16* sBl = sm + 3 * MT * STRH;

    const int tid = threadIdx.x;
    const int wgN = blockIdx.x;          // 0..1023
    const int wgM = blockIdx.y;          // 0..7

    // ---- async-stage A (q hi/lo) and B (queue hi/lo) blocks into LDS ----
    {
        const _Float16* gAh = qh  + (size_t)wgM * MT * D_;
        const _Float16* gAl = ql  + (size_t)wgM * MT * D_;
        const _Float16* gBh = bhv + (size_t)wgN * NT * D_;
        const _Float16* gBl = blv + (size_t)wgN * NT * D_;
        for (int idx = tid; idx < MT * (D_ / 8); idx += 256) {
            const int r = idx >> 5;
            const int c = (idx & 31) * 8;                 // 8 halves = 16 B
            const int gOff = (r * D_ + c) * 2;            // global byte offset
            async_copy16(gAh, gOff, (unsigned)(uintptr_t)(sAh + r * STRH + c));
            async_copy16(gAl, gOff, (unsigned)(uintptr_t)(sAl + r * STRH + c));
            async_copy16(gBh, gOff, (unsigned)(uintptr_t)(sBh + r * STRH + c));
            async_copy16(gBl, gOff, (unsigned)(uintptr_t)(sBl + r * STRH + c));
        }
        asm volatile("s_wait_asynccnt 0x0" ::: "memory");
    }
    __syncthreads();

    const int lane = tid & 31;
    const int wave = tid >> 5;
    const int mw   = wave >> 1;        // 0..3
    const int nw   = wave & 1;         // 0..1
    const int lm   = lane & 15;
    const int hi   = lane >> 4;        // 0 | 1
    const int aRow  = mw * 16 + lm;
    const int aKoff = hi * 8;          // 16-bit A layout: lanes16-31 start at K=8
    const int bKoff = hi * 8;          // 16-bit B layout: lanes16-31 start at K=8 (same as A)

    v8f acc0 = {}; v8f acc1 = {};

    for (int ks = 0; ks < D_ / 32; ++ks) {
        const int kb = ks * 32;
        const v16h ah = cat16(*(const v8h*)(sAh + aRow * STRH + kb + aKoff),
                              *(const v8h*)(sAh + aRow * STRH + kb + 16 + aKoff));
        const v16h al = cat16(*(const v8h*)(sAl + aRow * STRH + kb + aKoff),
                              *(const v8h*)(sAl + aRow * STRH + kb + 16 + aKoff));
#pragma unroll
        for (int t2 = 0; t2 < 2; ++t2) {
            const int n = nw * 32 + t2 * 16 + lm;
            const v16h bh = cat16(*(const v8h*)(sBh + n * STRH + kb + bKoff),
                                  *(const v8h*)(sBh + n * STRH + kb + 16 + bKoff));
            const v16h bl = cat16(*(const v8h*)(sBl + n * STRH + kb + bKoff),
                                  *(const v8h*)(sBl + n * STRH + kb + 16 + bKoff));
            v8f acc = t2 ? acc1 : acc0;
            acc = wmma_f16(ah, bh, acc);   // hi*hi
            acc = wmma_f16(ah, bl, acc);   // hi*lo
            acc = wmma_f16(al, bh, acc);   // lo*hi   (lo*lo ~ 2^-44, dropped)
            if (t2) acc1 = acc; else acc0 = acc;
        }
    }

    // ---- C/D layout: VGPR r -> M = r + 8*hi ; N = lane&15 ----
#pragma unroll
    for (int t2 = 0; t2 < 2; ++t2) {
        const v8f acc = t2 ? acc1 : acc0;
        const int col = wgN * NT + nw * 32 + t2 * 16 + lm;
#pragma unroll
        for (int r = 0; r < 8; ++r) {
            const int rowg = wgM * MT + mw * 16 + r + hi * 8;
            logits[(size_t)rowg * LROW + NPOS_ + col] = acc[r] * INV_T;
        }
    }
}

// =====================================================================
// K5: per-row stats over score_neg (recovered as logits*T, L2-warm)
// =====================================================================
__global__ void k_stats(const float* __restrict__ logits,
                        const float* __restrict__ score_pos,
                        const float* __restrict__ pos_min,
                        float* __restrict__ score)
{
    __shared__ float red[32];
    const int b = blockIdx.x, t = threadIdx.x;
    const float* row = logits + (size_t)b * LROW + NPOS_;
    const float pmin = pos_min[b];
    float s = 0.f, ss = 0.f, mx = -1e30f, cnt = 0.f;
    for (int i = t; i < KQ; i += 256) {
        const float v = row[i] * T_;
        s += v; ss += v * v;
        mx = fmaxf(mx, v);
        cnt += (v > pmin) ? 1.f : 0.f;
    }
    for (int o = 16; o > 0; o >>= 1) {
        s  += __shfl_down(s,  o, 32);
        ss += __shfl_down(ss, o, 32);
        mx  = fmaxf(mx, __shfl_down(mx, o, 32));
        cnt += __shfl_down(cnt, o, 32);
    }
    const int lane = t & 31, w = t >> 5;
    if (lane == 0) { red[w] = s; red[8 + w] = ss; red[16 + w] = mx; red[24 + w] = cnt; }
    __syncthreads();
    if (t == 0) {
        float S = 0.f, SS = 0.f, MX = -1e30f, CN = 0.f;
        for (int i = 0; i < 8; ++i) {
            S += red[i]; SS += red[8 + i];
            MX = fmaxf(MX, red[16 + i]); CN += red[24 + i];
        }
        const float mean = S / (float)KQ;
        const float var  = (SS - (float)KQ * mean * mean) / (float)(KQ - 1);
        const float sp0 = score_pos[b * 3 + 0];
        const float sp1 = score_pos[b * 3 + 1];
        const float sp2 = score_pos[b * 3 + 2];
        const float cp = (MX > sp0 ? 1.f : 0.f) + (MX > sp1 ? 1.f : 0.f)
                       + (MX > sp2 ? 1.f : 0.f);
        float* o = score + (size_t)b * 7;
        o[0] = mean; o[1] = var; o[2] = CN; o[3] = cp;
        o[4] = sp0;  o[5] = sp1; o[6] = sp2;
    }
}

// =====================================================================
// K6: labels = ones  (128 MB; non-temporal so logits stay L2-resident)
// =====================================================================
__global__ void k_labels(float* __restrict__ labels)
{
    const size_t n = (size_t)B_ * LROW;
    const size_t stride = (size_t)gridDim.x * blockDim.x * 4;
    const v4f ones = {1.f, 1.f, 1.f, 1.f};
    for (size_t i = ((size_t)blockIdx.x * blockDim.x + threadIdx.x) * 4;
         i < n; i += stride)
        __builtin_nontemporal_store(ones, (v4f*)(labels + i));
}

// =====================================================================
// K7: FIFO shift: new_queue[0 : K-3B] = queue[3B : K]  (streaming, NT)
// =====================================================================
__global__ void k_copyq(const float* __restrict__ queue, float* __restrict__ newq)
{
    const size_t n = (size_t)(KQ - NPOS_ * B_) * D_;
    const float* src = queue + (size_t)NPOS_ * B_ * D_;
    const size_t stride = (size_t)gridDim.x * blockDim.x * 4;
    for (size_t i = ((size_t)blockIdx.x * blockDim.x + threadIdx.x) * 4;
         i < n; i += stride) {
        const v4f v = __builtin_nontemporal_load((const v4f*)(src + i));
        __builtin_nontemporal_store(v, (v4f*)(newq + i));
    }
}

// =====================================================================
extern "C" void kernel_launch(void* const* d_in, const int* in_sizes, int n_in,
                              void* d_out, int out_size, void* d_ws, size_t ws_size,
                              hipStream_t stream)
{
    (void)in_sizes; (void)n_in; (void)out_size; (void)ws_size;

    const float* query   = (const float*)d_in[0];
    const float* key_emb = (const float*)d_in[1];
    const float* queue   = (const float*)d_in[2];

    float* out    = (float*)d_out;
    char*  ws     = (char*)d_ws;

    float*    q_norm    = (float*)   (ws + WS_QNORM);
    _Float16* qh        = (_Float16*)(ws + WS_QH);
    _Float16* ql        = (_Float16*)(ws + WS_QL);
    float*    score_pos = (float*)   (ws + WS_SPOS);
    float*    pos_min   = (float*)   (ws + WS_PMIN);
    _Float16* bhv       = (_Float16*)(ws + WS_BH);
    _Float16* blv       = (_Float16*)(ws + WS_BL);

    float* logits = out + OUT_LOGITS;
    float* labels = out + OUT_LABELS;
    float* score  = out + OUT_SCORE;
    float* newq   = out + OUT_NEWQ;

    k_normalize  <<<B_ + NPOS_ * B_, 256, 0, stream>>>(query, key_emb, q_norm, qh, ql, newq);
    k_split_queue<<<4096, 256, 0, stream>>>(queue, bhv, blv);
    k_pos        <<<B_, 256, 0, stream>>>(q_norm, newq, logits, score_pos, pos_min);

    dim3 gg(KQ / NT, B_ / MT);                       // 1024 x 8 workgroups
    const size_t lds_bytes = (size_t)4 * MT * STRH * sizeof(_Float16);  // 132 KB
    k_gemm       <<<gg, 256, lds_bytes, stream>>>(qh, ql, bhv, blv, logits);

    k_stats      <<<B_, 256, 0, stream>>>(logits, score_pos, pos_min, score);
    k_labels     <<<4096, 256, 0, stream>>>(labels);
    k_copyq      <<<4096, 256, 0, stream>>>(queue, newq);
}
